// SparsemaxAttention_90142773608636
// MI455X (gfx1250) — matrix operand order repaired and checked by reading, not compiled, not run on hardware
//
#include <hip/hip_runtime.h>
#include <hip/hip_bf16.h>

// ---------------------------------------------------------------------------
// Sparsemax attention, MI455X (gfx1250, wave32, WMMA).
//   B=2, N=2048, C=512, H=8, D=64
// Pipeline:
//   1) fp32 -> bf16 converts (x, Wq, Wk, Wv, Wo)
//   2) Q = x@Wq^T, K = x@Wk^T            -> bf16 [B,H,N,D]
//      V = x@Wv^T                        -> bf16 [B,H,D,N]  (transposed!)
//   3) scores = Q@K^T / 8  written fp32 straight into attn slice of d_out
//   4) sparsemax in place (exact, bisection on tau, no sort; wave32 shuffles)
//   5) O = attn@V: attn fp32 -> bf16 on the fly; V tile double-buffered in
//      LDS via global_load_async_to_lds_b128 (ASYNCcnt pipeline)
//   6) out = O@Wo^T + bo  -> fp32 into d_out
// All matrix math via v_wmma_f32_16x16x32_bf16 (fp32 accumulate).
// ---------------------------------------------------------------------------

typedef __bf16 bf16;
typedef __attribute__((ext_vector_type(8)))  __bf16 v8bf;
typedef __attribute__((ext_vector_type(16))) __bf16 v16bf;
typedef __attribute__((ext_vector_type(8)))  float  v8f;
typedef __attribute__((ext_vector_type(4)))  float  v4f;

static constexpr int Bb  = 2;
static constexpr int Nn  = 2048;
static constexpr int Cc  = 512;
static constexpr int Hh  = 8;
static constexpr int Dd  = 64;
static constexpr int BN  = Bb * Nn;        // 4096 rows for the dense GEMMs
static constexpr int BH  = Bb * Hh;        // 16 (b,h) pairs

// ---------------- fragment loaders (NT-form GEMM: A[M,K] * Bw[N,K]^T) -------

// A 16x32 bf16 fragment, row-major source. ISA layout:
//  lane m = lane&15; half = lane>>4; elems 0..7 = K=k0+8h.., elems 8..15 = K=k0+16+8h..
__device__ __forceinline__ v16bf load_a_bf16(const bf16* __restrict__ p,
                                             int mbase, int ld, int k0, int lane) {
  const int m = mbase + (lane & 15);
  const int half = lane >> 4;
  const bf16* base = p + (size_t)m * ld + k0 + 8 * half;
  v8bf lo = *(const v8bf*)(base);
  v8bf hi = *(const v8bf*)(base + 16);
  v16bf a;
#pragma unroll
  for (int i = 0; i < 8; ++i) { a[i] = lo[i]; a[i + 8] = hi[i]; }
  return a;
}

// Same A fragment but from fp32 source (attn matrix), converted to bf16.
__device__ __forceinline__ v16bf load_a_f32(const float* __restrict__ p,
                                            int mbase, int ld, int k0, int lane) {
  const int m = mbase + (lane & 15);
  const int half = lane >> 4;
  const float* b0 = p + (size_t)m * ld + k0 + 8 * half;
  const float* b1 = b0 + 16;
  v4f x0 = *(const v4f*)(b0);
  v4f x1 = *(const v4f*)(b0 + 4);
  v4f x2 = *(const v4f*)(b1);
  v4f x3 = *(const v4f*)(b1 + 4);
  v16bf a;
#pragma unroll
  for (int i = 0; i < 4; ++i) {
    a[i]      = (bf16)x0[i];
    a[i + 4]  = (bf16)x1[i];
    a[i + 8]  = (bf16)x2[i];
    a[i + 12] = (bf16)x3[i];
  }
  return a;
}

// B 32x16 bf16 fragment; source stored [Nout, K] row-major (so B^T load is
// contiguous along K). ISA layout: lane n = lane&15; half = lane>>4;
// elems 0..7 = K=k0+8h.., elems 8..15 = K=k0+16+8h..
__device__ __forceinline__ v16bf load_b_bf16(const bf16* __restrict__ p,
                                             int nbase, int ld, int k0, int lane) {
  const int n = nbase + (lane & 15);
  const int half = lane >> 4;
  const bf16* base = p + (size_t)n * ld + k0 + 8 * half;
  v8bf lo = *(const v8bf*)(base);
  v8bf hi = *(const v8bf*)(base + 16);
  v16bf b;
#pragma unroll
  for (int i = 0; i < 8; ++i) { b[i] = lo[i]; b[i + 8] = hi[i]; }
  return b;
}

__device__ __forceinline__ v8f wmma_bf16(v16bf a, v16bf b, v8f c) {
  return __builtin_amdgcn_wmma_f32_16x16x32_bf16(false, a, false, b, (short)0, c,
                                                 false, false);
}

// ---------------- kernel 1: fp32 -> bf16 convert ----------------------------
__global__ void f2bf_kernel(const float* __restrict__ src, bf16* __restrict__ dst,
                            int n) {
  int i = blockIdx.x * blockDim.x + threadIdx.x;
  const int stride = gridDim.x * blockDim.x;
  for (; i < n; i += stride) dst[i] = (bf16)src[i];
}

// ---------------- kernel 2/6: projection GEMM  y = A @ W^T + bias -----------
// A: bf16 [4096, 512]; W: bf16 [512, 512] ([out, in]); each wave -> 16x64 tile.
// MODE 0: bf16 out, head-split [B,H,N,D]             (Q, K)
// MODE 1: bf16 out, head-split transposed [B,H,D,N]  (V)
// MODE 2: fp32 out, row-major [4096, 512]            (final projection)
template <int MODE>
__global__ void proj_gemm_kernel(const bf16* __restrict__ A,
                                 const bf16* __restrict__ W,
                                 const float* __restrict__ bias,
                                 void* __restrict__ outp) {
  const int lane = threadIdx.x & 31;
  const int wid  = blockIdx.x * (blockDim.x >> 5) + (threadIdx.x >> 5);
  const int ngrp = wid & 7;        // 512 / 64 = 8 column groups
  const int mt   = wid >> 3;       // 256 row tiles
  const int mbase = mt * 16;
  const int nbase = ngrp * 64;

  v8f acc[4];
#pragma unroll
  for (int j = 0; j < 4; ++j)
#pragma unroll
    for (int r = 0; r < 8; ++r) acc[j][r] = 0.0f;

  for (int k0 = 0; k0 < Cc; k0 += 32) {
    v16bf a = load_a_bf16(A, mbase, Cc, k0, lane);
    __builtin_prefetch(A + (size_t)(mbase + (lane & 15)) * Cc + k0 + 32, 0, 1);
#pragma unroll
    for (int j = 0; j < 4; ++j) {
      v16bf b = load_b_bf16(W, nbase + 16 * j, Cc, k0, lane);
      acc[j] = wmma_bf16(a, b, acc[j]);
    }
  }

  const int nlane = lane & 15;
  const int half  = lane >> 4;
#pragma unroll
  for (int j = 0; j < 4; ++j) {
    const int col = nbase + 16 * j + nlane;
    const float bv = bias[col];
#pragma unroll
    for (int r = 0; r < 8; ++r) {
      const int m = mbase + r + 8 * half;
      const float val = acc[j][r] + bv;
      if constexpr (MODE == 0) {
        const int b_ = m >> 11, nseq = m & (Nn - 1);
        const int h = col >> 6, d = col & 63;
        ((bf16*)outp)[(((size_t)(b_ * Hh + h)) * Nn + nseq) * Dd + d] = (bf16)val;
      } else if constexpr (MODE == 1) {
        const int b_ = m >> 11, nseq = m & (Nn - 1);
        const int h = col >> 6, d = col & 63;
        ((bf16*)outp)[(((size_t)(b_ * Hh + h)) * Dd + d) * Nn + nseq] = (bf16)val;
      } else {
        ((float*)outp)[(size_t)m * Cc + col] = val;
      }
    }
  }
}

// ---------------- kernel 3: scores = Q @ K^T / sqrt(D) ----------------------
// Per (b,h): Q,K bf16 [2048,64]; out fp32 [2048,2048] in d_out attn slice.
__global__ void scores_kernel(const bf16* __restrict__ Q,
                              const bf16* __restrict__ Km,
                              float* __restrict__ attn) {
  const int lane = threadIdx.x & 31;
  const int wid  = blockIdx.x * (blockDim.x >> 5) + (threadIdx.x >> 5);
  const int bh  = wid >> 12;          // 4096 waves per (b,h)
  const int rem = wid & 4095;
  const int ngrp = rem & 31;          // 2048/64
  const int mt   = rem >> 5;          // 2048/16
  const bf16* q = Q  + (size_t)bh * Nn * Dd;
  const bf16* k = Km + (size_t)bh * Nn * Dd;
  float* o = attn + (size_t)bh * Nn * Nn;
  const int mbase = mt * 16, nbase = ngrp * 64;

  v8f acc[4];
#pragma unroll
  for (int j = 0; j < 4; ++j)
#pragma unroll
    for (int r = 0; r < 8; ++r) acc[j][r] = 0.0f;

#pragma unroll
  for (int k0 = 0; k0 < Dd; k0 += 32) {
    v16bf a = load_a_bf16(q, mbase, Dd, k0, lane);
#pragma unroll
    for (int j = 0; j < 4; ++j) {
      v16bf b = load_b_bf16(k, nbase + 16 * j, Dd, k0, lane);
      acc[j] = wmma_bf16(a, b, acc[j]);
    }
  }

  const int nlane = lane & 15;
  const int half  = lane >> 4;
  const float scale = 0.125f;  // 1/sqrt(64)
#pragma unroll
  for (int j = 0; j < 4; ++j) {
    const int col = nbase + 16 * j + nlane;
#pragma unroll
    for (int r = 0; r < 8; ++r) {
      const int m = mbase + r + 8 * half;
      o[(size_t)m * Nn + col] = acc[j][r] * scale;
    }
  }
}

// ---------------- kernel 4: sparsemax, in place on attn rows ----------------
// Exact without sorting: f(tau)=sum(max(z-tau,0)) is piecewise-linear,
// strictly decreasing; bisect on [zmax-1, zmax] to pin the support set,
// then tau = (sum_support - 1)/k exactly. Wave32 shuffle reductions keep it
// to 2 barriers per bisection step.
__device__ __forceinline__ float wave_reduce_add(float v) {
#pragma unroll
  for (int off = 16; off > 0; off >>= 1) v += __shfl_xor(v, off, 32);
  return v;
}
__device__ __forceinline__ float wave_reduce_max(float v) {
#pragma unroll
  for (int off = 16; off > 0; off >>= 1) v = fmaxf(v, __shfl_xor(v, off, 32));
  return v;
}

__global__ void sparsemax_kernel(float* __restrict__ attn) {
  __shared__ float wred[8];
  const int tid  = threadIdx.x;
  const int lane = tid & 31;
  const int wv   = tid >> 5;
  float* z = attn + (size_t)blockIdx.x * Nn;

  float v[8];
#pragma unroll
  for (int i = 0; i < 8; ++i) v[i] = z[tid + 256 * i];

  // row max
  float m = v[0];
#pragma unroll
  for (int i = 1; i < 8; ++i) m = fmaxf(m, v[i]);
  m = wave_reduce_max(m);
  if (lane == 0) wred[wv] = m;
  __syncthreads();
  float zmax = wred[0];
#pragma unroll
  for (int i = 1; i < 8; ++i) zmax = fmaxf(zmax, wred[i]);
  __syncthreads();

  float lo = zmax - 1.0f, hi = zmax;  // f(lo) >= 1 > f(hi) = 0
  for (int it = 0; it < 30; ++it) {
    const float mid = 0.5f * (lo + hi);
    float s = 0.0f;
#pragma unroll
    for (int i = 0; i < 8; ++i) s += fmaxf(v[i] - mid, 0.0f);
    s = wave_reduce_add(s);
    if (lane == 0) wred[wv] = s;
    __syncthreads();
    float tot = 0.0f;
#pragma unroll
    for (int i = 0; i < 8; ++i) tot += wred[i];
    __syncthreads();
    if (tot > 1.0f) lo = mid; else hi = mid;
  }

  // support statistics with threshold lo
  float cnt = 0.0f, sum = 0.0f;
#pragma unroll
  for (int i = 0; i < 8; ++i) {
    if (v[i] > lo) { cnt += 1.0f; sum += v[i]; }
  }
  cnt = wave_reduce_add(cnt);
  sum = wave_reduce_add(sum);
  if (lane == 0) wred[wv] = cnt;
  __syncthreads();
  float kk = 0.0f;
#pragma unroll
  for (int i = 0; i < 8; ++i) kk += wred[i];
  __syncthreads();
  if (lane == 0) wred[wv] = sum;
  __syncthreads();
  float ss = 0.0f;
#pragma unroll
  for (int i = 0; i < 8; ++i) ss += wred[i];
  __syncthreads();

  const float tau = (ss - 1.0f) / kk;
#pragma unroll
  for (int i = 0; i < 8; ++i) z[tid + 256 * i] = fmaxf(v[i] - tau, 0.0f);
}

// ---------------- kernel 5: O = attn @ V ------------------------------------
// Per (b,h): attn fp32 [2048,2048] (converted to bf16 per-fragment),
// Vt bf16 [64, 2048] (NT form). All 8 waves of a block share one (b,h), so
// the 64x32 V chunk per K-step is staged once into LDS with
// global_load_async_to_lds_b128 (double-buffered, ASYNCcnt-fenced) instead of
// being re-read from L2 by every wave. Output bf16 into [B,N,C] at col h*64.
__global__ void attnv_kernel(const float* __restrict__ attn,
                             const bf16* __restrict__ Vt,
                             bf16* __restrict__ O) {
  // two buffers of 64 rows x 32 bf16 (4 KB each)
  __shared__ __align__(16) bf16 bbuf[2][64 * 32];

  const int tid  = threadIdx.x;
  const int lane = tid & 31;
  const int wid  = blockIdx.x * (blockDim.x >> 5) + (tid >> 5);
  const int bh = wid >> 7;            // 128 row tiles per (b,h); 16 blocks/bh
  const int mt = wid & 127;
  const float* ap = attn + (size_t)bh * Nn * Nn;
  const bf16* vt  = Vt + (size_t)bh * Dd * Nn;
  const int mbase = mt * 16;

  // this thread's async-copy slot: row d = tid/4, 16-byte segment tid%4
  const int cp_d   = tid >> 2;
  const int cp_seg = tid & 3;
  const bf16* cp_src = vt + (size_t)cp_d * Nn + 8 * cp_seg;

  auto issue_chunk = [&](int buf, int k0) {
    unsigned lds = (unsigned)(uintptr_t)(&bbuf[buf][cp_d * 32 + 8 * cp_seg]);
    unsigned long long ga = (unsigned long long)(uintptr_t)(cp_src + k0);
    asm volatile("global_load_async_to_lds_b128 %0, %1, off"
                 :: "v"(lds), "v"(ga) : "memory");
  };

  v8f acc[4];
#pragma unroll
  for (int j = 0; j < 4; ++j)
#pragma unroll
    for (int r = 0; r < 8; ++r) acc[j][r] = 0.0f;

  issue_chunk(0, 0);

  const int half = lane >> 4;
  for (int k0 = 0; k0 < Nn; k0 += 32) {
    const int cur = (k0 >> 5) & 1;
    asm volatile("s_wait_asynccnt 0x0" ::: "memory");
    __syncthreads();                      // chunk [cur] visible to all waves
    if (k0 + 32 < Nn) issue_chunk(cur ^ 1, k0 + 32);

    v16bf a = load_a_f32(ap, mbase, Nn, k0, lane);
    __builtin_prefetch(ap + (size_t)(mbase + (lane & 15)) * Nn + k0 + 32, 0, 1);
#pragma unroll
    for (int j = 0; j < 4; ++j) {
      // B fragment from LDS: col d = 16j + lane&15, elems = 8*half + 0..7, 16 + 8*half + 0..7
      const bf16* bp = &bbuf[cur][(16 * j + (lane & 15)) * 32 + 8 * half];
      v8bf lo = *(const v8bf*)(bp);
      v8bf hi = *(const v8bf*)(bp + 16);
      v16bf b;
#pragma unroll
      for (int i = 0; i < 8; ++i) { b[i] = lo[i]; b[i + 8] = hi[i]; }
      acc[j] = wmma_bf16(a, b, acc[j]);
    }
    __syncthreads();                      // all reads of [cur] done before reuse
  }

  const int b_ = bh >> 3, h = bh & 7;
  const int nlane = lane & 15;
#pragma unroll
  for (int j = 0; j < 4; ++j) {
    const int d = 16 * j + nlane;
#pragma unroll
    for (int r = 0; r < 8; ++r) {
      const int nseq = mbase + r + 8 * half;
      O[((size_t)(b_ * Nn + nseq)) * Cc + h * Dd + d] = (bf16)acc[j][r];
    }
  }
}

// ---------------------------------------------------------------------------
extern "C" void kernel_launch(void* const* d_in, const int* in_sizes, int n_in,
                              void* d_out, int out_size, void* d_ws, size_t ws_size,
                              hipStream_t stream) {
  const float* x  = (const float*)d_in[0];
  const float* Wq = (const float*)d_in[1];
  const float* bq = (const float*)d_in[2];
  const float* Wk = (const float*)d_in[3];
  const float* bk = (const float*)d_in[4];
  const float* Wv = (const float*)d_in[5];
  const float* bv = (const float*)d_in[6];
  const float* Wo = (const float*)d_in[7];
  const float* bo = (const float*)d_in[8];

  float* out  = (float*)d_out;                       // [B,N,C]  = 2,097,152 f32
  float* attn = out + (size_t)BN * Cc;               // [B,H,N,N]= 67,108,864 f32

  // workspace carve-up (bf16), ~22 MiB total
  bf16* xbf = (bf16*)d_ws;                           // 4096*512
  bf16* wqb = xbf + (size_t)BN * Cc;                 // 512*512 each
  bf16* wkb = wqb + (size_t)Cc * Cc;
  bf16* wvb = wkb + (size_t)Cc * Cc;
  bf16* wob = wvb + (size_t)Cc * Cc;
  bf16* qbf = wob + (size_t)Cc * Cc;                 // [B,H,N,D]
  bf16* kbf = qbf + (size_t)BH * Nn * Dd;            // [B,H,N,D]
  bf16* vtb = kbf + (size_t)BH * Nn * Dd;            // [B,H,D,N]
  bf16* obf = vtb + (size_t)BH * Nn * Dd;            // [B,N,C]

  // 1) converts
  f2bf_kernel<<<512, 256, 0, stream>>>(x,  xbf, BN * Cc);
  f2bf_kernel<<<256, 256, 0, stream>>>(Wq, wqb, Cc * Cc);
  f2bf_kernel<<<256, 256, 0, stream>>>(Wk, wkb, Cc * Cc);
  f2bf_kernel<<<256, 256, 0, stream>>>(Wv, wvb, Cc * Cc);
  f2bf_kernel<<<256, 256, 0, stream>>>(Wo, wob, Cc * Cc);

  // 2) projections: 2048 waves each -> 256 blocks x 256 threads (8 waves)
  proj_gemm_kernel<0><<<256, 256, 0, stream>>>(xbf, wqb, bq, qbf);
  proj_gemm_kernel<0><<<256, 256, 0, stream>>>(xbf, wkb, bk, kbf);
  proj_gemm_kernel<1><<<256, 256, 0, stream>>>(xbf, wvb, bv, vtb);

  // 3) scores -> attn slice of d_out: 65536 waves -> 8192 blocks
  scores_kernel<<<8192, 256, 0, stream>>>(qbf, kbf, attn);

  // 4) sparsemax in place: one block per row
  sparsemax_kernel<<<BH * Nn, 256, 0, stream>>>(attn);

  // 5) attn @ V: 2048 waves -> 256 blocks (async-LDS staged V)
  attnv_kernel<<<256, 256, 0, stream>>>(attn, vtb, obf);

  // 6) final projection -> fp32 out
  proj_gemm_kernel<2><<<256, 256, 0, stream>>>(obf, wob, bo, out);
}
